// RAGGNN_V3_69767448756414
// MI455X (gfx1250) — matrix operand order repaired and checked by reading, not compiled, not run on hardware
//
#include <hip/hip_runtime.h>

typedef __bf16 bf16;
typedef __attribute__((ext_vector_type(8)))  bf16  v8bf;
typedef __attribute__((ext_vector_type(16))) bf16  v16bf;
typedef __attribute__((ext_vector_type(8)))  float v8f;
typedef __attribute__((ext_vector_type(4)))  float v4f;

union V16U { v16bf v; v8bf h[2]; };

// ---------------------------------------------------------------------------
// WMMA helpers (CDNA5 wave32, 16x16x32 bf16 -> f32)
// A-frag 16x32 (MxK): lane L holds row L&15; VGPR0-3 = K {0..7 | 8..15},
// VGPR4-7 = K {16..23 | 24..31} depending on lane half -> two b128 loads.
// B-frag 32x16 (KxN): lane L holds col L&15, same K order as A-frag.
// ---------------------------------------------------------------------------
__device__ __forceinline__ v8f wmma_bf16(v16bf a, v16bf b, v8f c) {
  return __builtin_amdgcn_wmma_f32_16x16x32_bf16(false, a, false, b, (short)0,
                                                 c, false, false);
}

__device__ __forceinline__ v16bf load_frag_a(const bf16* base, int ld, int lane) {
  const bf16* p = base + (lane & 15) * ld + ((lane >> 4) << 3);
  V16U u;
  u.h[0] = *(const v8bf*)(p);
  u.h[1] = *(const v8bf*)(p + 16);
  return u.v;
}

__device__ __forceinline__ v16bf load_frag_b(const bf16* base, int ld, int lane) {
  const bf16* p = base + (lane & 15) * ld + ((lane >> 4) << 3);
  V16U u;
  u.h[0] = *(const v8bf*)(p);
  u.h[1] = *(const v8bf*)(p + 16);
  return u.v;
}

__device__ __forceinline__ float gelu_f(float x) {
  return 0.5f * x * (1.0f + erff(x * 0.70710678118654752440f));
}

// ---------------------------------------------------------------------------
// Prep kernels
// ---------------------------------------------------------------------------
// vectorized f32 -> bf16 (n, nsrc multiples of 8; [nsrc, n) zero-filled)
__global__ void cvt_bf16x8_kernel(const float* __restrict__ src,
                                  bf16* __restrict__ dst, size_t n,
                                  size_t nsrc) {
  size_t i = ((size_t)blockIdx.x * blockDim.x + threadIdx.x) * 8;
  if (i >= n) return;
  v8bf o;
  if (i < nsrc) {
    v4f a = *(const v4f*)(src + i);
    v4f b = *(const v4f*)(src + i + 4);
#pragma unroll
    for (int k = 0; k < 4; ++k) {
      o[k] = (bf16)a[k];
      o[k + 4] = (bf16)b[k];
    }
  } else {
#pragma unroll
    for (int k = 0; k < 8; ++k) o[k] = (bf16)0.0f;
  }
  *(v8bf*)(dst + i) = o;
}

// X [8192x128] f32 -> XT [128x8192] bf16
__global__ void xt_kernel(const float* __restrict__ X, bf16* __restrict__ XT) {
  int idx = blockIdx.x * 256 + threadIdx.x;  // h*8192 + n
  int n = idx & 8191, h = idx >> 13;
  XT[idx] = (bf16)X[(size_t)n * 128 + h];
}

// ---------------------------------------------------------------------------
// GNN layer: out = GELU(LN((A @ Hin) @ W^T))
// A bf16 [N x N], HT bf16 [H x N] (input transposed), W bf16 [H x H] row-major.
// Block = 64 rows, 4 waves x (16 rows x 128 cols); grid = 128 for machine fill.
// K staged through LDS in 128-wide chunks.
// ---------------------------------------------------------------------------
__global__ __launch_bounds__(128) void gnn_layer_kernel(
    const bf16* __restrict__ Abf, const bf16* __restrict__ HT,
    const bf16* __restrict__ W, const float* __restrict__ lng,
    const float* __restrict__ lnb, bf16* __restrict__ outT,
    float* __restrict__ outF, bf16* __restrict__ outBF) {
  const int N = 8192, H = 128;
  __shared__ __align__(16) bf16 sB[128 * 136];  // staged B chunk [128][128+8]
  __shared__ __align__(16) bf16 sC[64 * 136];   // C tile (A@H) [64][128+8]
  const int tid = threadIdx.x;
  const int lane = tid & 31;
  const int w = tid >> 5;  // 0..3
  const int ln16 = lane & 15;
  const int lh = lane >> 4;
  const int row0 = blockIdx.x * 64;

  const v8f zero = {0.f, 0.f, 0.f, 0.f, 0.f, 0.f, 0.f, 0.f};
  v8f acc[8];
#pragma unroll
  for (int t = 0; t < 8; ++t) acc[t] = zero;

  for (int k0 = 0; k0 < N; k0 += 128) {
    __syncthreads();
#pragma unroll
    for (int u16 = 0; u16 < 16; ++u16) {
      int u = tid * 16 + u16;  // 2048 v8bf units: [row 0..127][part 0..15]
      int rh = u >> 4, part = u & 15;
      *(v8bf*)(sB + rh * 136 + part * 8) =
          *(const v8bf*)(HT + (size_t)rh * N + k0 + part * 8);
    }
    __syncthreads();
#pragma unroll
    for (int kk = 0; kk < 128; kk += 32) {
      v16bf a = load_frag_a(Abf + (size_t)(row0 + w * 16) * N + k0 + kk, N, lane);
#pragma unroll
      for (int t = 0; t < 8; ++t) {
        v16bf b = load_frag_b(sB + t * 16 * 136 + kk, 136, lane);
        acc[t] = wmma_bf16(a, b, acc[t]);
      }
    }
  }

  // (A@H) tile -> LDS as bf16 (D layout: row = w*16 + p + 8*lh, col = t*16+ln16)
#pragma unroll
  for (int t = 0; t < 8; ++t)
#pragma unroll
    for (int p = 0; p < 8; ++p)
      sC[(w * 16 + p + lh * 8) * 136 + t * 16 + ln16] = (bf16)acc[t][p];
  __syncthreads();

  // second pass: tile @ W^T  (W row-major [out_j][k] -> B-frag contiguous in k)
  v8f acc2[8];
#pragma unroll
  for (int t = 0; t < 8; ++t) acc2[t] = zero;
#pragma unroll
  for (int ks = 0; ks < 128; ks += 32) {
    v16bf a = load_frag_a(sC + w * 16 * 136 + ks, 136, lane);
#pragma unroll
    for (int t = 0; t < 8; ++t) {
      v16bf b = load_frag_b(W + t * 16 * H + ks, H, lane);
      acc2[t] = wmma_bf16(a, b, acc2[t]);
    }
  }

  float gv[8], bv[8];
#pragma unroll
  for (int t = 0; t < 8; ++t) {
    gv[t] = lng[t * 16 + ln16];
    bv[t] = lnb[t * 16 + ln16];
  }

  // LayerNorm over H=128 per row + GELU. Row p's 128 cols live in acc2[0..7][p]
  // across the 16 lanes of this lane-half -> xor-shuffle reduce (masks 1..8).
#pragma unroll
  for (int p = 0; p < 8; ++p) {
    float s = 0.f, q = 0.f;
#pragma unroll
    for (int t = 0; t < 8; ++t) {
      float x = acc2[t][p];
      s += x;
      q += x * x;
    }
#pragma unroll
    for (int m = 1; m < 16; m <<= 1) {
      s += __shfl_xor(s, m, 32);
      q += __shfl_xor(q, m, 32);
    }
    float mean = s * (1.f / 128.f);
    float var = q * (1.f / 128.f) - mean * mean;
    float rstd = rsqrtf(var + 1e-5f);
#pragma unroll
    for (int t = 0; t < 8; ++t)
      acc2[t][p] = gelu_f((acc2[t][p] - mean) * rstd * gv[t] + bv[t]);
  }

  // outputs: transposed bf16 (next layer B operand), optional f32/bf16 row-major
  const int nodeb = row0 + w * 16 + lh * 8;
#pragma unroll
  for (int t = 0; t < 8; ++t) {
    int col = t * 16 + ln16;
    v8bf pk;
#pragma unroll
    for (int p = 0; p < 8; ++p) pk[p] = (bf16)acc2[t][p];
    *(v8bf*)(outT + (size_t)col * N + nodeb) = pk;  // 8 consecutive nodes
    if (outF) {
#pragma unroll
      for (int p = 0; p < 8; ++p)
        outF[(size_t)(nodeb + p) * H + col] = acc2[t][p];
    }
    if (outBF) {
#pragma unroll
      for (int p = 0; p < 8; ++p)
        outBF[(size_t)(nodeb + p) * H + col] = (bf16)acc2[t][p];
    }
  }
}

// ---------------------------------------------------------------------------
// Projection: q = GELU(gnn @ pw1^T + b1) @ pw2^T + b2   -> q bf16 [N x 64]
// ---------------------------------------------------------------------------
__global__ __launch_bounds__(256) void proj_kernel(
    const bf16* __restrict__ gnnBF, const bf16* __restrict__ PW1,
    const float* __restrict__ b1, const bf16* __restrict__ PW2,
    const float* __restrict__ b2, bf16* __restrict__ qBF) {
  const int H = 128, D = 64;
  __shared__ __align__(16) bf16 sC[128 * 136];
  const int tid = threadIdx.x;
  const int lane = tid & 31;
  const int w = tid >> 5;
  const int ln16 = lane & 15;
  const int lh = lane >> 4;
  const int row0 = blockIdx.x * 128;

  const v8f zero = {0.f, 0.f, 0.f, 0.f, 0.f, 0.f, 0.f, 0.f};
  v8f acc[8];
#pragma unroll
  for (int t = 0; t < 8; ++t) acc[t] = zero;
#pragma unroll
  for (int ks = 0; ks < 128; ks += 32) {
    v16bf a = load_frag_a(gnnBF + (size_t)(row0 + w * 16) * H + ks, H, lane);
#pragma unroll
    for (int t = 0; t < 8; ++t) {
      v16bf b = load_frag_b(PW1 + t * 16 * H + ks, H, lane);
      acc[t] = wmma_bf16(a, b, acc[t]);
    }
  }
#pragma unroll
  for (int t = 0; t < 8; ++t) {
    float bb = b1[t * 16 + ln16];
#pragma unroll
    for (int p = 0; p < 8; ++p)
      sC[(w * 16 + p + lh * 8) * 136 + t * 16 + ln16] =
          (bf16)gelu_f(acc[t][p] + bb);
  }
  __syncthreads();

  v8f acc2[4];
#pragma unroll
  for (int t = 0; t < 4; ++t) acc2[t] = zero;
#pragma unroll
  for (int ks = 0; ks < 128; ks += 32) {
    v16bf a = load_frag_a(sC + w * 16 * 136 + ks, 136, lane);
#pragma unroll
    for (int t = 0; t < 4; ++t) {
      v16bf b = load_frag_b(PW2 + t * 16 * H + ks, H, lane);
      acc2[t] = wmma_bf16(a, b, acc2[t]);
    }
  }
#pragma unroll
  for (int t = 0; t < 4; ++t) {
    float bb = b2[t * 16 + ln16];
#pragma unroll
    for (int p = 0; p < 8; ++p)
      qBF[(size_t)(row0 + w * 16 + p + lh * 8) * D + t * 16 + ln16] =
          (bf16)(acc2[t][p] + bb);
  }
}

// ---------------------------------------------------------------------------
// Retrieval: scores (WMMA) -> top-10 -> softmax(T=0.5, incl. 1/sqrt(64)) -> ctx
// Block = 64 nodes, 4 waves; grid = 128. Doc chunks of 96 are staged in LDS
// (shared by all waves), scored into LDS, scanned by 2 threads/node.
// ---------------------------------------------------------------------------
#define RET_CH 96
#define RET_MPAD 20256  // 211 * 96, zero-padded docs

__global__ __launch_bounds__(128) void retrieve_kernel(
    const bf16* __restrict__ qBF, const bf16* __restrict__ docBF,
    const float* __restrict__ docF, float* __restrict__ ctx) {
  const int D = 64, M = 20000;
  __shared__ __align__(16) float sc[64 * 101];   // score tile
  __shared__ __align__(16) bf16 sDoc[96 * 72];   // staged doc chunk [96][64+8]
  const int tid = threadIdx.x;
  const int lane = tid & 31;
  const int w = tid >> 5;  // 0..3
  const int ln16 = lane & 15;
  const int lh = lane >> 4;
  const int row0 = blockIdx.x * 64;
  const v8f zero = {0.f, 0.f, 0.f, 0.f, 0.f, 0.f, 0.f, 0.f};

  // loop-invariant q fragments (K = 64 -> two k-steps)
  v16bf a0 = load_frag_a(qBF + (size_t)(row0 + w * 16) * D + 0, D, lane);
  v16bf a1 = load_frag_a(qBF + (size_t)(row0 + w * 16) * D + 32, D, lane);

  float tv[10];
  int ti[10];
#pragma unroll
  for (int i = 0; i < 10; ++i) {
    tv[i] = -1.0e30f;
    ti[i] = 0;
  }
  const int nodeL = tid & 63;
  const int half = tid >> 6;

  for (int mb = 0; mb < RET_MPAD; mb += RET_CH) {
    __syncthreads();  // prior scan / doc-frag reads done
    // stage doc chunk: 96 rows x 64 halves = 768 v8bf units, 6 per thread
#pragma unroll
    for (int u6 = 0; u6 < 6; ++u6) {
      int u = tid * 6 + u6;
      int rh = u >> 3, part = u & 7;
      *(v8bf*)(sDoc + rh * 72 + part * 8) =
          *(const v8bf*)(docBF + (size_t)(mb + rh) * D + part * 8);
    }
    __syncthreads();
#pragma unroll
    for (int t = 0; t < 6; ++t) {
      v16bf b0 = load_frag_b(sDoc + t * 16 * 72 + 0, 72, lane);
      v16bf b1 = load_frag_b(sDoc + t * 16 * 72 + 32, 72, lane);
      v8f c = wmma_bf16(a1, b1, wmma_bf16(a0, b0, zero));
#pragma unroll
      for (int p = 0; p < 8; ++p)
        sc[(w * 16 + p + lh * 8) * 101 + t * 16 + ln16] = c[p];
    }
    __syncthreads();
    for (int c0 = half * 48; c0 < half * 48 + 48; ++c0) {
      int m = mb + c0;
      if (m < M) {
        float s = sc[nodeL * 101 + c0];
        if (s > tv[0]) {  // replace current min, one bubble pass
          tv[0] = s;
          ti[0] = m;
#pragma unroll
          for (int j = 0; j < 9; ++j) {
            if (tv[j] > tv[j + 1]) {
              float xv = tv[j]; tv[j] = tv[j + 1]; tv[j + 1] = xv;
              int xi = ti[j]; ti[j] = ti[j + 1]; ti[j + 1] = xi;
            }
          }
        }
      }
    }
  }
  __syncthreads();

  // merge the two half-lists per node via LDS (reuse score buffer)
  float* mv = sc;
  int* mi = (int*)(sc + 1280);
  if (half) {
#pragma unroll
    for (int i = 0; i < 10; ++i) {
      mv[nodeL * 10 + i] = tv[i];
      mi[nodeL * 10 + i] = ti[i];
    }
  }
  __syncthreads();
  if (!half) {
#pragma unroll
    for (int i = 0; i < 10; ++i) {
      float s = mv[nodeL * 10 + i];
      int m = mi[nodeL * 10 + i];
      if (s > tv[0]) {
        tv[0] = s;
        ti[0] = m;
#pragma unroll
        for (int j = 0; j < 9; ++j) {
          if (tv[j] > tv[j + 1]) {
            float xv = tv[j]; tv[j] = tv[j + 1]; tv[j + 1] = xv;
            int xi = ti[j]; ti[j] = ti[j + 1]; ti[j + 1] = xi;
          }
        }
      }
    }
    // softmax: logits = (dot/8)/0.5 = dot*0.25 ; tv[9] is max
    float mx = tv[9];
    float aw[10], wsum = 0.f;
#pragma unroll
    for (int i = 0; i < 10; ++i) {
      aw[i] = expf((tv[i] - mx) * 0.25f);
      wsum += aw[i];
    }
    float inv = 1.f / wsum;
    int node = row0 + nodeL;
    for (int d = 0; d < D; ++d) {
      float c = 0.f;
#pragma unroll
      for (int i = 0; i < 10; ++i) c += aw[i] * docF[(size_t)ti[i] * D + d];
      ctx[(size_t)node * D + d] = c * inv;
    }
  }
}

// ---------------------------------------------------------------------------
// Gate fuse: g = sigmoid([gnn,ctx] @ gate_w^T + gb); out = g*gnn + (1-g)*(ctx@cw^T+cb)
// ---------------------------------------------------------------------------
__global__ __launch_bounds__(256) void fuse_kernel(
    const float* __restrict__ gnn, const float* __restrict__ ctx,
    const float* __restrict__ gw, const float* __restrict__ gb,
    const float* __restrict__ cw, const float* __restrict__ cb,
    float* __restrict__ out) {
  const int j = threadIdx.x & 127;
  const int node = (blockIdx.x << 1) + (threadIdx.x >> 7);
  const float* gn = gnn + (size_t)node * 128;
  const float* cx = ctx + (size_t)node * 64;
  const float* wg = gw + (size_t)j * 192;
  float a = gb[j];
#pragma unroll 4
  for (int k = 0; k < 128; ++k) a += wg[k] * gn[k];
#pragma unroll 4
  for (int k = 0; k < 64; ++k) a += wg[128 + k] * cx[k];
  float g = 1.f / (1.f + expf(-a));
  const float* wc = cw + (size_t)j * 64;
  float ct = cb[j];
#pragma unroll 4
  for (int k = 0; k < 64; ++k) ct += wc[k] * cx[k];
  out[(size_t)node * 128 + j] = g * gn[j] + (1.f - g) * ct;
}

// ---------------------------------------------------------------------------
// Host-side launch
// ---------------------------------------------------------------------------
extern "C" void kernel_launch(void* const* d_in, const int* in_sizes, int n_in,
                              void* d_out, int out_size, void* d_ws,
                              size_t ws_size, hipStream_t stream) {
  (void)in_sizes; (void)n_in; (void)out_size;
  const int N = 8192, H = 128, D = 64, M = 20000;

  const float* A = (const float*)d_in[0];
  const float* X = (const float*)d_in[1];
  const float* doc = (const float*)d_in[2];
  const float* W1 = (const float*)d_in[3];
  const float* W2 = (const float*)d_in[4];
  const float* W3 = (const float*)d_in[5];
  const float* ln1g = (const float*)d_in[6];
  const float* ln1b = (const float*)d_in[7];
  const float* ln2g = (const float*)d_in[8];
  const float* ln2b = (const float*)d_in[9];
  const float* ln3g = (const float*)d_in[10];
  const float* ln3b = (const float*)d_in[11];
  const float* pw1 = (const float*)d_in[12];
  const float* pb1 = (const float*)d_in[13];
  const float* pw2 = (const float*)d_in[14];
  const float* pb2 = (const float*)d_in[15];
  const float* gw = (const float*)d_in[16];
  const float* gb = (const float*)d_in[17];
  const float* cw = (const float*)d_in[18];
  const float* cb = (const float*)d_in[19];

  char* ws = (char*)d_ws;
  size_t off = 0;
  auto carve = [&](size_t bytes) {
    void* p = ws + off;
    off = (off + bytes + 255) & ~(size_t)255;
    return p;
  };
  bf16* Abf  = (bf16*)carve((size_t)N * N * 2);
  bf16* HTa  = (bf16*)carve((size_t)H * N * 2);
  bf16* HTb  = (bf16*)carve((size_t)H * N * 2);
  float* gnnF = (float*)carve((size_t)N * H * 4);
  bf16* gnnB = (bf16*)carve((size_t)N * H * 2);
  bf16* qB   = (bf16*)carve((size_t)N * D * 2);
  bf16* docB = (bf16*)carve((size_t)RET_MPAD * D * 2);
  float* ctxF = (float*)carve((size_t)N * D * 4);
  bf16* w1B  = (bf16*)carve((size_t)H * H * 2);
  bf16* w2B  = (bf16*)carve((size_t)H * H * 2);
  bf16* w3B  = (bf16*)carve((size_t)H * H * 2);
  bf16* pw1B = (bf16*)carve((size_t)H * H * 2);
  bf16* pw2B = (bf16*)carve((size_t)D * H * 2);
  if (off > ws_size) return;  // scratch too small: bail (deterministic no-op)

  // --- prep: bf16 conversions / transposes ---
  {
    size_t n = (size_t)N * N;  // 64M elems, 8 per thread
    cvt_bf16x8_kernel<<<(int)(n / 8 / 256), 256, 0, stream>>>(A, Abf, n, n);
  }
  xt_kernel<<<(N * H) / 256, 256, 0, stream>>>(X, HTa);
  cvt_bf16x8_kernel<<<8, 256, 0, stream>>>(W1, w1B, H * H, H * H);
  cvt_bf16x8_kernel<<<8, 256, 0, stream>>>(W2, w2B, H * H, H * H);
  cvt_bf16x8_kernel<<<8, 256, 0, stream>>>(W3, w3B, H * H, H * H);
  cvt_bf16x8_kernel<<<8, 256, 0, stream>>>(pw1, pw1B, H * H, H * H);
  cvt_bf16x8_kernel<<<4, 256, 0, stream>>>(pw2, pw2B, D * H, D * H);
  {
    size_t n = (size_t)RET_MPAD * D, ns = (size_t)M * D;
    cvt_bf16x8_kernel<<<(int)((n / 8 + 255) / 256), 256, 0, stream>>>(doc, docB,
                                                                      n, ns);
  }

  // --- 3-layer GNN (WMMA bf16, f32 accumulate) ---
  gnn_layer_kernel<<<N / 64, 128, 0, stream>>>(Abf, HTa, w1B, ln1g, ln1b, HTb,
                                               nullptr, nullptr);
  gnn_layer_kernel<<<N / 64, 128, 0, stream>>>(Abf, HTb, w2B, ln2g, ln2b, HTa,
                                               nullptr, nullptr);
  gnn_layer_kernel<<<N / 64, 128, 0, stream>>>(Abf, HTa, w3B, ln3g, ln3b, HTb,
                                               gnnF, gnnB);

  // --- projection to doc space ---
  proj_kernel<<<N / 128, 256, 0, stream>>>(gnnB, pw1B, pb1, pw2B, pb2, qB);

  // --- retrieval: WMMA scores + top-k + softmax-weighted gather ---
  retrieve_kernel<<<N / 64, 128, 0, stream>>>(qB, docB, doc, ctxF);

  // --- sigmoid gate fuse ---
  fuse_kernel<<<N / 2, 256, 0, stream>>>(gnnF, ctxF, gw, gb, cw, cb,
                                         (float*)d_out);
}
